// PraxisAttention_5901285064840
// MI455X (gfx1250) — matrix operand order repaired and checked by reading, not compiled, not run on hardware
//
#include <hip/hip_runtime.h>
#include <hip/hip_bf16.h>
#include <math.h>

// ---------------------------------------------------------------------------
// PraxisAttention for MI455X (gfx1250): bf16 WMMA pipeline
//   x[4096,1024] -> Q,K,V proj -> ALiBi causal flash-attention -> O proj
// GEMMs: 4-wave workgroups, 128x64 tile, B staged in LDS via async-to-LDS.
// ---------------------------------------------------------------------------

typedef __bf16 bf16_t;
typedef __attribute__((ext_vector_type(16))) __bf16 v16bf;
typedef __attribute__((ext_vector_type(8)))  __bf16 v8bf;
typedef __attribute__((ext_vector_type(8)))  float  v8f;
typedef __attribute__((ext_vector_type(4)))  int    v4i;

#define S_LEN   4096
#define D_MODEL 1024
#define NHEADS  16
#define HDIM    64

#define BM      128    // rows per workgroup (4 waves x 32 rows)
#define BN      64     // cols per workgroup
#define KSTEP   32     // K per WMMA step
#define KPAD    40     // padded LDS row stride (bf16) to spread banks

#if defined(__gfx1250__) && __has_builtin(__builtin_amdgcn_global_load_async_to_lds_b128)
#define HAVE_ASYNC_LDS 1
#endif

// Build a 16-element bf16 A/B fragment from two 16-byte chunks (global or LDS).
__device__ __forceinline__ v16bf load_frag16(const bf16_t* p0, const bf16_t* p1) {
    union { v16bf v; v8bf h[2]; } u;
    u.h[0] = *(const v8bf*)p0;
    u.h[1] = *(const v8bf*)p1;
    return u.v;
}

__device__ __forceinline__ v8f wmma_bf16(v16bf a, v16bf b, v8f c) {
    return __builtin_amdgcn_wmma_f32_16x16x32_bf16(false, a, false, b,
                                                   (short)0, c, false, false);
}

// 16-byte global -> LDS copy, async (ASYNCcnt) when available.
// Builtin prototype (clang-22/gfx1250): void(v4i* AS1, v4i* AS3, imm, imm).
__device__ __forceinline__ void async_ld_b128(bf16_t* lds_dst, const bf16_t* src) {
#ifdef HAVE_ASYNC_LDS
    v4i* g = (v4i*)(bf16_t*)src;   // drop const, retype
    v4i* l = (v4i*)lds_dst;
    __builtin_amdgcn_global_load_async_to_lds_b128(
        (__attribute__((address_space(1))) v4i*)g,
        (__attribute__((address_space(3))) v4i*)l,
        0, 0);
#else
    *(v8bf*)lds_dst = *(const v8bf*)src;
#endif
}

template <int N>
__device__ __forceinline__ void wait_async() {
#ifdef HAVE_ASYNC_LDS
#if __has_builtin(__builtin_amdgcn_s_wait_asynccnt)
    __builtin_amdgcn_s_wait_asynccnt(N);
#else
    asm volatile("s_wait_asynccnt %0" :: "i"(N) : "memory");
#endif
#endif
}

// ---------------------------------------------------------------------------
// fp32 -> bf16 staging (8 elements / thread, b128 loads)
// ---------------------------------------------------------------------------
__global__ void cvt_f32_bf16(const float* __restrict__ in,
                             bf16_t* __restrict__ out, int n8) {
    int i = blockIdx.x * blockDim.x + threadIdx.x;
    if (i >= n8) return;
    const float4* p = (const float4*)in + (size_t)i * 2;
    float4 a = p[0], b = p[1];
    v8bf o;
    o[0] = (__bf16)a.x; o[1] = (__bf16)a.y; o[2] = (__bf16)a.z; o[3] = (__bf16)a.w;
    o[4] = (__bf16)b.x; o[5] = (__bf16)b.y; o[6] = (__bf16)b.z; o[7] = (__bf16)b.w;
    *((v8bf*)out + i) = o;
}

// ---------------------------------------------------------------------------
// Staged GEMM: out[s,n] = sum_k X[s,k]*W[n,k] (+bias[n])
// Workgroup = 128 threads (4 waves), tile 128x64; each wave owns 32 rows.
// B (weight) tile double-buffered in LDS via async-to-LDS.
// mode 0: bf16 head-major [H][S][64]   (Q, K)
// mode 1: bf16 transposed [H][64][S]   (V)
// mode 2: f32  row-major  [S][D]       (final output)
// ---------------------------------------------------------------------------
__global__ void __launch_bounds__(128)
gemm_staged(const bf16_t* __restrict__ X, const bf16_t* __restrict__ W,
            const float* __restrict__ bias, bf16_t* __restrict__ outb,
            float* __restrict__ outf, int mode) {
    __shared__ __align__(16) bf16_t bbuf[2][BN * KPAD];   // 2 x 5 KB

    const int tid  = threadIdx.x;
    const int wave = tid >> 5;
    const int lane = tid & 31;
    const int half = lane >> 4;   // 0 or 1
    const int l16  = lane & 15;

    const int s_wave = blockIdx.x * BM + wave * 32;   // 32 rows per wave
    const int c0     = blockIdx.y * BN;

    // Issue the 64x32 B tile for k0 into buffer `buf`:
    // 256 16B chunks; each of the 128 threads copies 2.
    auto issueB = [&](int k0, int buf) {
#pragma unroll
        for (int c = 0; c < 2; ++c) {
            int ch   = tid + c * 128;        // 0..255
            int col  = ch >> 2;              // 0..63
            int part = ch & 3;               // 8 bf16 each
            async_ld_b128(&bbuf[buf][col * KPAD + part * 8],
                          W + (size_t)(c0 + col) * D_MODEL + k0 + part * 8);
        }
    };

    v8f acc[2][4] = {};
    const bf16_t* xrow0 = X + (size_t)(s_wave + l16) * D_MODEL;
    const bf16_t* xrow1 = X + (size_t)(s_wave + 16 + l16) * D_MODEL;

    issueB(0, 0);
    const int NK = D_MODEL / KSTEP;   // 32
    for (int ik = 0; ik < NK; ++ik) {
        const int k0 = ik * KSTEP;
        if (ik + 1 < NK) { issueB(k0 + KSTEP, (ik + 1) & 1); wait_async<2>(); }
        else             { wait_async<0>(); }
        __syncthreads();   // buffer ik&1 visible to all waves

        // A fragments for this wave's two 16-row blocks (direct from L2)
        v16bf a0 = load_frag16(xrow0 + k0 + half * 8, xrow0 + k0 + half * 8 + 16);
        v16bf a1 = load_frag16(xrow1 + k0 + half * 8, xrow1 + k0 + half * 8 + 16);

        const bf16_t* bb = bbuf[ik & 1];
#pragma unroll
        for (int j = 0; j < 4; ++j) {
            const bf16_t* bp = bb + (j * 16 + l16) * KPAD + half * 8;
            v16bf b = load_frag16(bp, bp + 16);
            acc[0][j] = wmma_bf16(a0, b, acc[0][j]);
            acc[1][j] = wmma_bf16(a1, b, acc[1][j]);
        }
        __syncthreads();   // all reads done before next overwrite of this buffer
    }

    // ---- epilogue ----
    const int r0 = half * 8;
#pragma unroll
    for (int rb = 0; rb < 2; ++rb) {
#pragma unroll
        for (int j = 0; j < 4; ++j) {
            int col = c0 + j * 16 + l16;
            float bv = bias ? bias[col] : 0.0f;
            int h = col >> 6, hd = col & 63;
#pragma unroll
            for (int i = 0; i < 8; ++i) {
                int row = s_wave + rb * 16 + r0 + i;
                float v = acc[rb][j][i] + bv;
                if (mode == 2) {
                    outf[(size_t)row * D_MODEL + col] = v;
                } else if (mode == 1) {
                    outb[((size_t)h * HDIM + hd) * S_LEN + row] = (bf16_t)v;
                } else {
                    outb[((size_t)h * S_LEN + row) * HDIM + hd] = (bf16_t)v;
                }
            }
        }
    }
}

// ---------------------------------------------------------------------------
// Flash attention with ALiBi + causal mask.
// One wave per (head, 16-query tile); 32-key blocks up to the diagonal.
// Q,K head-major [H][S][64]; V transposed [H][64][S]; O bf16 [S][1024].
// ---------------------------------------------------------------------------
__global__ void __launch_bounds__(32)
attn_kernel(const bf16_t* __restrict__ Q, const bf16_t* __restrict__ Kh,
            const bf16_t* __restrict__ Vt, bf16_t* __restrict__ O) {
    __shared__ __align__(16) bf16_t pbuf[16 * 32];

    const int lane = threadIdx.x;
    const int half = lane >> 4;
    const int l16  = lane & 15;
    const int qs0  = blockIdx.x * 16;
    const int h    = blockIdx.y;

    const float slope = exp2f(-0.5f * (float)(h + 1));  // 2^(-8*(h+1)/16)

    const bf16_t* Qh = Q  + (size_t)h * S_LEN * HDIM;
    const bf16_t* Kb = Kh + (size_t)h * S_LEN * HDIM;
    const bf16_t* Vb = Vt + (size_t)h * HDIM * S_LEN;

    // Q as two A fragments (K-dim 0..31 and 32..63)
    const bf16_t* qrow = Qh + (size_t)(qs0 + l16) * HDIM;
    v16bf aq0 = load_frag16(qrow + half * 8,      qrow + half * 8 + 16);
    v16bf aq1 = load_frag16(qrow + 32 + half * 8, qrow + 32 + half * 8 + 16);

    v8f acc[4] = {};
    float m_i[8], l_i[8];
#pragma unroll
    for (int i = 0; i < 8; ++i) { m_i[i] = -1e30f; l_i[i] = 0.0f; }

    const int r0   = half * 8;
    const int kend = qs0 + 16;

    for (int kb0 = 0; kb0 < kend; kb0 += 32) {
        // ---- scores = Q @ K^T : 16x32 as two 16x16 D fragments ----
        v8f sc[2];
#pragma unroll
        for (int nt = 0; nt < 2; ++nt) {
            const bf16_t* krow = Kb + (size_t)(kb0 + nt * 16 + l16) * HDIM
                                    + half * 8;
            v16bf b0 = load_frag16(krow,      krow + 16);  // d = 0..31 chunk
            v16bf b1 = load_frag16(krow + 32, krow + 48);  // d = 32..63 chunk
            v8f c = {};
            c = wmma_bf16(aq0, b0, c);
            c = wmma_bf16(aq1, b1, c);
            sc[nt] = c;
        }

        // ---- scale, ALiBi, causal mask, online softmax ----
        float p0[8], p1[8];
        const int key0 = kb0 + l16;
        const int key1 = kb0 + 16 + l16;
#pragma unroll
        for (int i = 0; i < 8; ++i) {
            const int qrow_i = qs0 + r0 + i;
            float v0 = sc[0][i] * 0.125f - slope * (float)key0;
            float v1 = sc[1][i] * 0.125f - slope * (float)key1;
            v0 = (key0 <= qrow_i) ? v0 : -1e30f;
            v1 = (key1 <= qrow_i) ? v1 : -1e30f;
            float bm = fmaxf(v0, v1);
#pragma unroll
            for (int off = 1; off < 16; off <<= 1)
                bm = fmaxf(bm, __shfl_xor(bm, off, 16));
            float mn   = fmaxf(m_i[i], bm);
            float corr = __expf(m_i[i] - mn);
            float e0   = __expf(v0 - mn);
            float e1   = __expf(v1 - mn);
            float rs   = e0 + e1;
#pragma unroll
            for (int off = 1; off < 16; off <<= 1)
                rs += __shfl_xor(rs, off, 16);
            l_i[i] = l_i[i] * corr + rs;
            m_i[i] = mn;
            p0[i] = e0;
            p1[i] = e1;
#pragma unroll
            for (int j = 0; j < 4; ++j) acc[j][i] *= corr;
        }

        // ---- relayout P: D-fragment -> A-fragment through LDS ----
        __syncthreads();
#pragma unroll
        for (int i = 0; i < 8; ++i) {
            pbuf[(r0 + i) * 32 + l16]      = (bf16_t)p0[i];
            pbuf[(r0 + i) * 32 + 16 + l16] = (bf16_t)p1[i];
        }
        __syncthreads();
        v16bf pA = load_frag16(&pbuf[l16 * 32 + half * 8],
                               &pbuf[l16 * 32 + half * 8 + 16]);

        // ---- acc += P @ V (V transposed: contiguous along keys) ----
#pragma unroll
        for (int j = 0; j < 4; ++j) {
            const bf16_t* vrow = Vb + (size_t)(j * 16 + l16) * S_LEN
                                    + kb0 + half * 8;
            v16bf bv = load_frag16(vrow, vrow + 16);
            acc[j] = wmma_bf16(pA, bv, acc[j]);
        }
    }

    // ---- normalize and write O[s, h*64 + dv] as bf16 ----
#pragma unroll
    for (int j = 0; j < 4; ++j) {
        int col = h * HDIM + j * 16 + l16;
#pragma unroll
        for (int i = 0; i < 8; ++i) {
            int row = qs0 + r0 + i;
            float v = acc[j][i] / l_i[i];
            O[(size_t)row * D_MODEL + col] = (bf16_t)v;
        }
    }
}

// ---------------------------------------------------------------------------
// Host-side launcher
// ---------------------------------------------------------------------------
extern "C" void kernel_launch(void* const* d_in, const int* in_sizes, int n_in,
                              void* d_out, int out_size, void* d_ws, size_t ws_size,
                              hipStream_t stream) {
    const float* x  = (const float*)d_in[0];
    const float* Wq = (const float*)d_in[1];
    const float* bq = (const float*)d_in[2];
    const float* Wk = (const float*)d_in[3];
    const float* Wv = (const float*)d_in[4];
    const float* bv = (const float*)d_in[5];
    const float* Wo = (const float*)d_in[6];
    float* out = (float*)d_out;

    const size_t SD = (size_t)S_LEN * D_MODEL;   // 4M elems
    const size_t DD = (size_t)D_MODEL * D_MODEL; // 1M elems

    bf16_t* ws  = (bf16_t*)d_ws;
    bf16_t* xb  = ws;  ws += SD;
    bf16_t* wqb = ws;  ws += DD;
    bf16_t* wkb = ws;  ws += DD;
    bf16_t* wvb = ws;  ws += DD;
    bf16_t* wob = ws;  ws += DD;
    bf16_t* qb  = ws;  ws += SD;   // head-major [H][S][64]
    bf16_t* kb  = ws;  ws += SD;   // head-major [H][S][64]
    bf16_t* vtb = ws;  ws += SD;   // transposed [H][64][S]
    bf16_t* ob  = ws;  ws += SD;   // attn out   [S][1024]

    auto cvt = [&](const float* in, bf16_t* o, size_t n) {
        int n8 = (int)(n / 8);
        cvt_f32_bf16<<<(n8 + 255) / 256, 256, 0, stream>>>(in, o, n8);
    };
    cvt(x,  xb,  SD);
    cvt(Wq, wqb, DD);
    cvt(Wk, wkb, DD);
    cvt(Wv, wvb, DD);
    cvt(Wo, wob, DD);

    dim3 gg(S_LEN / BM, D_MODEL / BN);   // (32, 16)
    gemm_staged<<<gg, 128, 0, stream>>>(xb, wqb, bq,      qb,  nullptr, 0);
    gemm_staged<<<gg, 128, 0, stream>>>(xb, wkb, nullptr, kb,  nullptr, 0);
    gemm_staged<<<gg, 128, 0, stream>>>(xb, wvb, bv,      vtb, nullptr, 1);

    dim3 ga(S_LEN / 16, NHEADS);
    attn_kernel<<<ga, 32, 0, stream>>>(qb, kb, vtb, ob);

    gemm_staged<<<gg, 128, 0, stream>>>(ob, wob, nullptr, nullptr, out, 2);
}
